// CWGDN_41394894799830
// MI455X (gfx1250) — matrix operand order repaired and checked by reading, not compiled, not run on hardware
//
#include <hip/hip_runtime.h>
#include <hip/hip_bf16.h>

// ---------------------------------------------------------------------------
// CDNA5 (gfx1250) implementation of the gated dynamic-conv block.
// Heavy GEMMs (temporal pointwise convs) run on v_wmma_f32_16x16x32_f16,
// with A tiles staged into LDS via async global->LDS loads (ASYNCcnt).
// ---------------------------------------------------------------------------

typedef __attribute__((ext_vector_type(16))) _Float16 v16h;
typedef __attribute__((ext_vector_type(8)))  _Float16 v8h;
typedef __attribute__((ext_vector_type(8)))  float    v8f;

#define BT   16          // b*t
#define TT   8
#define CIN  64
#define HID  128
#define HH   128
#define WW2  128
#define HWP  (HH*WW2)    // 16384
#define LNEPS 1e-5f

#if defined(__gfx1250__) && __has_builtin(__builtin_amdgcn_global_load_async_to_lds_b128) && __has_builtin(__builtin_amdgcn_s_wait_asynccnt)
#define HAS_ASYNC 1
#else
#define HAS_ASYNC 0
#endif

// Parameter type of the async builtins in this toolchain (from diagnostics):
// generic pointer to a 16-byte int vector.
typedef int b128_t __attribute__((vector_size(4 * sizeof(int))));

// ---------------------------------------------------------------------------
// 1) LayerNorm over 64 channels per pixel; emit f16, position-major (p,hw,c).
// ---------------------------------------------------------------------------
__global__ __launch_bounds__(256) void k_ln(const float* __restrict__ x,
                                            const float* __restrict__ lw,
                                            const float* __restrict__ lb,
                                            _Float16* __restrict__ xnT) {
  int idx = blockIdx.x * blockDim.x + threadIdx.x;      // p*HW + hw
  if (idx >= BT * HWP) return;
  int p  = idx >> 14;
  int hw = idx & (HWP - 1);
  const float* bp = x + (size_t)p * CIN * HWP + hw;
  float s = 0.f, s2 = 0.f;
  for (int c = 0; c < CIN; ++c) {
    float v = bp[(size_t)c * HWP];
    s += v; s2 += v * v;
  }
  float mu  = s * (1.0f / CIN);
  float var = s2 * (1.0f / CIN) - mu * mu;
  float inv = __frsqrt_rn(var + LNEPS);
  _Float16* dst = xnT + ((size_t)idx << 6);             // *64, contiguous channels
  for (int c0 = 0; c0 < CIN; c0 += 8) {
    v8h pack;
#pragma unroll
    for (int j = 0; j < 8; ++j) {
      int c = c0 + j;
      float v = bp[(size_t)c * HWP];
      pack[j] = (_Float16)((v - mu) * inv * lw[c] + lb[c]);
    }
    *(v8h*)(dst + c0) = pack;
  }
}

// ---------------------------------------------------------------------------
// 2) Weight prepack to f16, layout [kd][oc][ic] (K contiguous per row).
// ---------------------------------------------------------------------------
__global__ __launch_bounds__(256) void k_pack(const float* __restrict__ pin,
                                              const float* __restrict__ pout,
                                              _Float16* __restrict__ wpin,
                                              _Float16* __restrict__ wpout) {
  int i = blockIdx.x * blockDim.x + threadIdx.x;
  if (i < 3 * 256 * CIN) {             // pin_w: (256, 64, 3) -> [kd][oc][ic]
    int kd = i / (256 * CIN);
    int r  = i % (256 * CIN);
    int oc = r / CIN, ic = r % CIN;
    wpin[i] = (_Float16)pin[((size_t)oc * CIN + ic) * 3 + kd];
  }
  if (i < 3 * CIN * HID) {             // pout_w: (64, 128, 3) -> [kd][oc][ic]
    int kd = i / (CIN * HID);
    int r  = i % (CIN * HID);
    int oc = r / HID, ic = r % HID;
    wpout[i] = (_Float16)pout[((size_t)oc * HID + ic) * 3 + kd];
  }
}

// ---------------------------------------------------------------------------
// 3) pin temporal GEMM: y[p,oc,hw] = sum_kd sum_ic xn[p+kd-1,ic,hw]*W
//    M=16 pixels/block, N=256 (8 waves x 2 tiles), K = 3 x 64 (2 steps of 32).
//    A tiles (3 x 2KB, contiguous) staged to LDS via async loads.
// ---------------------------------------------------------------------------
__global__ __launch_bounds__(256) void k_pin(const _Float16* __restrict__ xnT,
                                             const _Float16* __restrict__ wpin,
                                             _Float16* __restrict__ x1h,
                                             _Float16* __restrict__ x2h) {
  __shared__ __align__(16) _Float16 smA[3][16 * CIN];   // 6 KB
  __shared__ float sm[8][16][17];
  int bid = blockIdx.x;                // 16 * 1024
  int p   = bid >> 10;
  int hw0 = (bid & 1023) << 4;
  int b   = p >> 3, t = p & 7;
  int tid  = threadIdx.x;
  int lane = tid & 31;
  int wv   = tid >> 5;                 // 0..7
  int nl   = lane & 15;                // A-row / B-col / D-col index
  int g    = lane >> 4;

  // Stage the three 16x64 f16 A tiles (each a contiguous 2KB block).
  for (int chunk = tid; chunk < 3 * 128; chunk += 256) {
    int kd = chunk >> 7, ci = chunk & 127;
    int tk = t + kd - 1;
    if (tk >= 0 && tk < TT) {
      int p2 = (b << 3) + tk;
      const _Float16* gsrc = xnT + ((size_t)(p2 * HWP + hw0) << 6) + ci * 8;
#if HAS_ASYNC
      __builtin_amdgcn_global_load_async_to_lds_b128(
          (b128_t*)gsrc, (b128_t*)&smA[kd][ci * 8], 0, 0);
#else
      *(v8h*)&smA[kd][ci * 8] = *(const v8h*)gsrc;
#endif
    }
  }
#if HAS_ASYNC
  __builtin_amdgcn_s_wait_asynccnt(0);
#endif
  __syncthreads();

  v8f acc0 = {}, acc1 = {};
  for (int kd = 0; kd < 3; ++kd) {
    int tk = t + kd - 1;
    if (tk < 0 || tk >= TT) continue;  // uniform per block
    const _Float16* wb = wpin + (size_t)kd * 256 * CIN;
    __builtin_prefetch(wb + (size_t)wv * 32 * CIN, 0, 1);
#pragma unroll
    for (int ks = 0; ks < 2; ++ks) {
      int k0 = ks * 32;
      v8h lo = *(const v8h*)&smA[kd][nl * CIN + k0 + 16 * g];
      v8h hi = *(const v8h*)&smA[kd][nl * CIN + k0 + 16 * g + 8];
      v16h a;
#pragma unroll
      for (int j = 0; j < 8; ++j) { a[j] = lo[j]; a[j + 8] = hi[j]; }
      int ocb = wv * 32;
      v16h b0 = *(const v16h*)(wb + (size_t)(ocb + nl) * CIN + k0 + 16 * g);
      v16h b1 = *(const v16h*)(wb + (size_t)(ocb + 16 + nl) * CIN + k0 + 16 * g);
      acc0 = __builtin_amdgcn_wmma_f32_16x16x32_f16(false, a, false, b0,
                                                    (short)0, acc0, false, false);
      acc1 = __builtin_amdgcn_wmma_f32_16x16x32_f16(false, a, false, b1,
                                                    (short)0, acc1, false, false);
    }
  }

  // LDS transpose per wave, then coalesced half stores into x1 / x2.
#pragma unroll
  for (int tile = 0; tile < 2; ++tile) {
    v8f acc = tile ? acc1 : acc0;
#pragma unroll
    for (int r = 0; r < 8; ++r) sm[wv][r + 8 * g][nl] = acc[r];  // sm[m][n]
    int ocbase = wv * 32 + tile * 16;
#pragma unroll
    for (int pass = 0; pass < 8; ++pass) {
      int nn = pass * 2 + g;
      int m  = nl;
      float v = sm[wv][m][nn];
      int ocg = ocbase + nn;
      _Float16* out = (ocg < HID) ? x1h : x2h;
      int c = ocg & (HID - 1);
      out[((size_t)p * HID + c) * HWP + hw0 + m] = (_Float16)v;
    }
  }
}

// ---------------------------------------------------------------------------
// 4) 2x2 average pool (128 -> 64), f16 in, f32 out.
// ---------------------------------------------------------------------------
__global__ __launch_bounds__(256) void k_avg(const _Float16* __restrict__ x1h,
                                             float* __restrict__ out) {
  int idx = blockIdx.x * blockDim.x + threadIdx.x;      // 16*128*64*64
  if (idx >= BT * HID * 64 * 64) return;
  int pc = idx >> 12;                 // (p*128+c)
  int q  = idx & 4095;
  int oh = q >> 6, ow = q & 63;
  const _Float16* src = x1h + (size_t)pc * HWP + (oh * 2) * WW2 + ow * 2;
  float v = (float)src[0] + (float)src[1] + (float)src[WW2] + (float)src[WW2 + 1];
  out[idx] = v * 0.25f;
}

// ---------------------------------------------------------------------------
// 5) depthwise 3x3 conv, SAME padding, f32.
// ---------------------------------------------------------------------------
__global__ __launch_bounds__(256) void k_dw(const float* __restrict__ in,
                                            const float* __restrict__ w,
                                            const float* __restrict__ bias,
                                            float* __restrict__ out, int S) {
  int idx = blockIdx.x * blockDim.x + threadIdx.x;
  int total = BT * HID * S * S;
  if (idx >= total) return;
  int q  = idx % (S * S);
  int pc = idx / (S * S);
  int c  = pc & (HID - 1);
  int oh = q / S, ow = q % S;
  const float* base = in + (size_t)pc * S * S;
  float acc = bias[c];
#pragma unroll
  for (int dy = -1; dy <= 1; ++dy) {
    int ih = oh + dy;
    if (ih < 0 || ih >= S) continue;
#pragma unroll
    for (int dx = -1; dx <= 1; ++dx) {
      int iw = ow + dx;
      if (iw < 0 || iw >= S) continue;
      acc += base[ih * S + iw] * w[c * 9 + (dy + 1) * 3 + (dx + 1)];
    }
  }
  out[idx] = acc;
}

// ---------------------------------------------------------------------------
// 6) 2x2 max pool (64 -> 32), f32.
// ---------------------------------------------------------------------------
__global__ __launch_bounds__(256) void k_max(const float* __restrict__ in,
                                             float* __restrict__ out) {
  int idx = blockIdx.x * blockDim.x + threadIdx.x;      // 16*128*32*32
  if (idx >= BT * HID * 32 * 32) return;
  int pc = idx >> 10;
  int q  = idx & 1023;
  int oh = q >> 5, ow = q & 31;
  const float* src = in + (size_t)pc * 4096 + (oh * 2) * 64 + ow * 2;
  float v = fmaxf(fmaxf(src[0], src[1]), fmaxf(src[64], src[65]));
  out[idx] = v;
}

// ---------------------------------------------------------------------------
// 7) global mean over 32x32 -> pooled (16,128).
// ---------------------------------------------------------------------------
__global__ __launch_bounds__(128) void k_mean(const float* __restrict__ in,
                                              float* __restrict__ pooled) {
  __shared__ float red[128];
  int pc  = blockIdx.x;                // 0..2047
  int tid = threadIdx.x;
  const float* base = in + (size_t)pc * 1024;
  float s = 0.f;
  for (int i = tid; i < 1024; i += 128) s += base[i];
  red[tid] = s;
  __syncthreads();
  for (int st = 64; st > 0; st >>= 1) {
    if (tid < st) red[tid] += red[tid + st];
    __syncthreads();
  }
  if (tid == 0) pooled[pc] = red[0] * (1.0f / 1024.0f);
}

// ---------------------------------------------------------------------------
// 8) token GEMM: kern[p][j] = pooled[p] . tok_w[j] + tok_b[j]   (16 x 1152)
// ---------------------------------------------------------------------------
__global__ __launch_bounds__(256) void k_tok(const float* __restrict__ pooled,
                                             const float* __restrict__ tw,
                                             const float* __restrict__ tb,
                                             float* __restrict__ kern) {
  int idx = blockIdx.x * blockDim.x + threadIdx.x;
  if (idx >= BT * 1152) return;
  int p = idx / 1152, j = idx % 1152;
  const float* pv = pooled + p * HID;
  const float* wr = tw + (size_t)j * HID;
  float s = tb[j];
  for (int c = 0; c < HID; ++c) s += pv[c] * wr[c];
  kern[idx] = s;
}

// ---------------------------------------------------------------------------
// 9) dynamic depthwise 3x3 + bias + gate; emit gated f16 position-major.
// ---------------------------------------------------------------------------
__global__ __launch_bounds__(128) void k_dyn(const _Float16* __restrict__ x1h,
                                             const _Float16* __restrict__ x2h,
                                             const float* __restrict__ kern,
                                             const float* __restrict__ dwb,
                                             _Float16* __restrict__ gated) {
  int bid = blockIdx.x;                // 16*128 blocks: (p, h)
  int p = bid >> 7, h = bid & 127;
  int c = threadIdx.x;                 // 0..127
  float kw[9];
  const float* kr = kern + ((size_t)p * HID + c) * 9;
#pragma unroll
  for (int j = 0; j < 9; ++j) kw[j] = kr[j];
  float bia = dwb[c];
  const _Float16* plane = x1h + ((size_t)p * HID + c) * HWP;
  const _Float16* x2p   = x2h + ((size_t)p * HID + c) * HWP + h * WW2;
  _Float16* gp = gated + ((size_t)(p * HWP + h * WW2)) * HID + c;
  for (int w = 0; w < WW2; ++w) {
    float acc = bia;
#pragma unroll
    for (int dy = -1; dy <= 1; ++dy) {
      int ih = h + dy;
      if (ih < 0 || ih >= HH) continue;
#pragma unroll
      for (int dx = -1; dx <= 1; ++dx) {
        int iw = w + dx;
        if (iw < 0 || iw >= WW2) continue;
        acc += (float)plane[ih * WW2 + iw] * kw[(dy + 1) * 3 + (dx + 1)];
      }
    }
    float gv = acc * (float)x2p[w];
    gp[(size_t)w * HID] = (_Float16)gv;
  }
}

// ---------------------------------------------------------------------------
// 10) pout temporal GEMM + residual: M=16 px/block, N=64 (4 waves), K=3x128.
//     A tiles (3 x 4KB, contiguous) staged to LDS via async loads.
// ---------------------------------------------------------------------------
__global__ __launch_bounds__(128) void k_pout(const _Float16* __restrict__ gated,
                                              const _Float16* __restrict__ wpout,
                                              const float* __restrict__ x,
                                              float* __restrict__ out) {
  __shared__ __align__(16) _Float16 smA[3][16 * HID];   // 12 KB
  __shared__ float sm[4][16][17];
  int bid  = blockIdx.x;               // 16 * 1024
  int p    = bid >> 10;
  int hw0  = (bid & 1023) << 4;
  int b    = p >> 3, t = p & 7;
  int tid  = threadIdx.x;
  int lane = tid & 31;
  int wv   = tid >> 5;                 // 0..3
  int nl   = lane & 15;
  int g    = lane >> 4;

  // Stage the three 16x128 f16 A tiles (each a contiguous 4KB block).
  for (int chunk = tid; chunk < 3 * 256; chunk += 128) {
    int kd = chunk >> 8, ci = chunk & 255;
    int tk = t + kd - 1;
    if (tk >= 0 && tk < TT) {
      int p2 = (b << 3) + tk;
      const _Float16* gsrc = gated + ((size_t)(p2 * HWP + hw0) << 7) + ci * 8;
#if HAS_ASYNC
      __builtin_amdgcn_global_load_async_to_lds_b128(
          (b128_t*)gsrc, (b128_t*)&smA[kd][ci * 8], 0, 0);
#else
      *(v8h*)&smA[kd][ci * 8] = *(const v8h*)gsrc;
#endif
    }
  }
#if HAS_ASYNC
  __builtin_amdgcn_s_wait_asynccnt(0);
#endif
  __syncthreads();

  v8f acc = {};
  for (int kd = 0; kd < 3; ++kd) {
    int tk = t + kd - 1;
    if (tk < 0 || tk >= TT) continue;  // uniform
    const _Float16* wb = wpout + (size_t)kd * CIN * HID;
    __builtin_prefetch(wb + (size_t)wv * 16 * HID, 0, 1);
#pragma unroll
    for (int ks = 0; ks < 4; ++ks) {
      int k0 = ks * 32;
      v8h lo = *(const v8h*)&smA[kd][nl * HID + k0 + 16 * g];
      v8h hi = *(const v8h*)&smA[kd][nl * HID + k0 + 16 * g + 8];
      v16h a;
#pragma unroll
      for (int j = 0; j < 8; ++j) { a[j] = lo[j]; a[j + 8] = hi[j]; }
      v16h bf = *(const v16h*)(wb + (size_t)(wv * 16 + nl) * HID + k0 + 16 * g);
      acc = __builtin_amdgcn_wmma_f32_16x16x32_f16(false, a, false, bf,
                                                   (short)0, acc, false, false);
    }
  }

#pragma unroll
  for (int r = 0; r < 8; ++r) sm[wv][r + 8 * g][nl] = acc[r];   // sm[m][n]
  int ocb = wv * 16;
#pragma unroll
  for (int pass = 0; pass < 8; ++pass) {
    int nn = pass * 2 + g;
    int m  = nl;
    int oc = ocb + nn;
    size_t oi = ((size_t)p * CIN + oc) * HWP + hw0 + m;
    out[oi] = sm[wv][m][nn] + x[oi];
  }
}

// ---------------------------------------------------------------------------
// Host side
// ---------------------------------------------------------------------------
extern "C" void kernel_launch(void* const* d_in, const int* in_sizes, int n_in,
                              void* d_out, int out_size, void* d_ws, size_t ws_size,
                              hipStream_t stream) {
  const float* x     = (const float*)d_in[0];
  const float* ln_w  = (const float*)d_in[1];
  const float* ln_b  = (const float*)d_in[2];
  const float* pin_w = (const float*)d_in[3];
  const float* pout_w= (const float*)d_in[4];
  const float* b1_w  = (const float*)d_in[5];
  const float* b1_b  = (const float*)d_in[6];
  const float* b2_w  = (const float*)d_in[7];
  const float* b2_b  = (const float*)d_in[8];
  const float* tok_w = (const float*)d_in[9];
  const float* tok_b = (const float*)d_in[10];
  const float* dw_b  = (const float*)d_in[11];
  float* out = (float*)d_out;

  // Workspace layout (bytes). 'gated' (67MB) overlays xnT+poolA, which are
  // dead by the time k_dyn runs. Total footprint ~252 MB.
  char* ws = (char*)d_ws;
  _Float16* xnT   = (_Float16*)(ws + 0);             //  33,554,432
  _Float16* gated = (_Float16*)(ws + 0);             //  67,108,864 (alias)
  float*    poolA = (float*)(ws + 33554432ull);      //  33,554,432
  float*    poolB = (float*)(ws + 67108864ull);      //  33,554,432
  float*    poolC1= (float*)(ws + 100663296ull);     //   8,388,608
  float*    poolC2= (float*)(ws + 109051904ull);     //   8,388,608
  _Float16* x1h   = (_Float16*)(ws + 117440512ull);  //  67,108,864
  _Float16* x2h   = (_Float16*)(ws + 184549376ull);  //  67,108,864
  float*    pooled= (float*)(ws + 251658240ull);     //       8,192
  float*    kern  = (float*)(ws + 251666432ull);     //      73,728
  _Float16* wpin  = (_Float16*)(ws + 251740160ull);  //      98,304
  _Float16* wpout = (_Float16*)(ws + 251838464ull);  //      49,152

  // 1) weight prepack + layernorm
  k_pack<<<(3 * 256 * CIN + 255) / 256, 256, 0, stream>>>(pin_w, pout_w, wpin, wpout);
  k_ln<<<(BT * HWP + 255) / 256, 256, 0, stream>>>(x, ln_w, ln_b, xnT);

  // 2) pin temporal GEMM (WMMA): -> x1h, x2h
  k_pin<<<BT * (HWP / 16), 256, 0, stream>>>(xnT, wpin, x1h, x2h);

  // 3) pooled branch
  k_avg<<<(BT * HID * 64 * 64 + 255) / 256, 256, 0, stream>>>(x1h, poolA);
  int n64 = BT * HID * 64 * 64, g64 = (n64 + 255) / 256;
  k_dw<<<g64, 256, 0, stream>>>(poolA, b1_w + 0 * HID * 9, b1_b + 0 * HID, poolB, 64);
  k_dw<<<g64, 256, 0, stream>>>(poolB, b1_w + 1 * HID * 9, b1_b + 1 * HID, poolA, 64);
  k_dw<<<g64, 256, 0, stream>>>(poolA, b1_w + 2 * HID * 9, b1_b + 2 * HID, poolB, 64);
  k_max<<<(BT * HID * 32 * 32 + 255) / 256, 256, 0, stream>>>(poolB, poolC1);
  int n32 = BT * HID * 32 * 32, g32 = (n32 + 255) / 256;
  k_dw<<<g32, 256, 0, stream>>>(poolC1, b2_w + 0 * HID * 9, b2_b + 0 * HID, poolC2, 32);
  k_dw<<<g32, 256, 0, stream>>>(poolC2, b2_w + 1 * HID * 9, b2_b + 1 * HID, poolC1, 32);
  k_dw<<<g32, 256, 0, stream>>>(poolC1, b2_w + 2 * HID * 9, b2_b + 2 * HID, poolC2, 32);
  k_mean<<<BT * HID, 128, 0, stream>>>(poolC2, pooled);
  k_tok<<<(BT * 1152 + 255) / 256, 256, 0, stream>>>(pooled, tok_w, tok_b, kern);

  // 4) dynamic depthwise + gating -> gated (f16, position-major)
  k_dyn<<<BT * HH, 128, 0, stream>>>(x1h, x2h, kern, dw_b, gated);

  // 5) pout temporal GEMM (WMMA) + residual -> out
  k_pout<<<BT * (HWP / 16), 128, 0, stream>>>(gated, wpout, x, out);
}
